// MultiheadChannelAttention_74655121539546
// MI455X (gfx1250) — matrix operand order repaired and checked by reading, not compiled, not run on hardware
//
#include <hip/hip_runtime.h>
#include <hip/hip_bf16.h>
#include <stdint.h>

typedef __bf16 bf16;
typedef __attribute__((ext_vector_type(16))) __bf16 v16bf;
typedef __attribute__((ext_vector_type(8)))  float  v8f;
typedef __attribute__((ext_vector_type(4)))  unsigned int v4u32;
typedef __attribute__((ext_vector_type(8)))  int          v8i32;
typedef __attribute__((ext_vector_type(4)))  int          v4i32;

// ---------------------------------------------------------------- helpers
__device__ __forceinline__ bf16 f2bf(float x) {
    unsigned u = __builtin_bit_cast(unsigned, x);
    unsigned r = (u + 0x7FFFu + ((u >> 16) & 1u)) >> 16;   // RNE
    unsigned short s = (unsigned short)r;
    return __builtin_bit_cast(bf16, s);
}

// ---------------------------------------------------------------- cast f32 -> bf16
__global__ void cast_kernel(const float* __restrict__ src, bf16* __restrict__ dst, size_t n) {
    size_t i = (size_t)blockIdx.x * blockDim.x + threadIdx.x;
    size_t stride = (size_t)gridDim.x * blockDim.x;
    for (; i < n; i += stride) dst[i] = f2bf(src[i]);
}

// ---------------------------------------------------------------- batched GEMM  C = A * B^T
// A: [M,K] rows K-contiguous (bf16), B: [N,K] rows K-contiguous (bf16)
// batch g = i*J + j ; operand offsets via (sAi,sAj)/(sBi,sBj); C contiguous per g (sCg elems).
// One wave (32 threads) per 64x64 C tile: 4x4 WMMA accumulators -> 16 WMMAs per
// 8 fragment loads per k-step (2.0 WMMA/load). No LDS, no barriers.
// Requires M,N multiples of 64 and K multiple of 32 (true for every call site here).
template <bool OUT_BF16>
__global__ __launch_bounds__(32) void gemm_bt(
    const bf16* __restrict__ A, const bf16* __restrict__ B, void* __restrict__ Cout,
    int M, int N, int K,
    long sAi, long sAj, long sBi, long sBj, long sCg, int J)
{
    int g  = blockIdx.z;
    int bi = g / J;
    int bj = g - bi * J;
    const bf16* Ag = A + (size_t)bi * sAi + (size_t)bj * sAj;
    const bf16* Bg = B + (size_t)bi * sBi + (size_t)bj * sBj;

    int lane = threadIdx.x & 31;
    int row0 = blockIdx.y * 64;
    int col0 = blockIdx.x * 64;

    int m = lane & 15;     // row-in-tile for A / col-in-tile for B
    int h = lane >> 4;     // lane half

    // A fragment (16x32): lane (h,m) holds k = {h*8..h*8+7} and {16+h*8..+7}
    // B fragment (32x16 col-major from N-major rows): lane (h,m)=(kh,n) holds k = {kh*8..kh*8+7} and {16+kh*8..+7}
    const bf16* pa[4];
    const bf16* pb[4];
#pragma unroll
    for (int t = 0; t < 4; t++) {
        pa[t] = Ag + (size_t)(row0 + t * 16 + m) * K + h * 8;
        pb[t] = Bg + (size_t)(col0 + t * 16 + m) * K + h * 8;
    }

    v8f acc[4][4] = {};

    for (int k0 = 0; k0 < K; k0 += 32) {
        v16bf a[4], b[4];
#pragma unroll
        for (int t = 0; t < 4; t++) {
            ((uint4*)&a[t])[0] = *(const uint4*)(pa[t] + k0);
            ((uint4*)&a[t])[1] = *(const uint4*)(pa[t] + k0 + 16);
            ((uint4*)&b[t])[0] = *(const uint4*)(pb[t] + k0);
            ((uint4*)&b[t])[1] = *(const uint4*)(pb[t] + k0 + 16);
        }
#pragma unroll
        for (int ti = 0; ti < 4; ti++)
#pragma unroll
            for (int tj = 0; tj < 4; tj++)
                acc[ti][tj] = __builtin_amdgcn_wmma_f32_16x16x32_bf16(
                    false, a[ti], false, b[tj], (short)0, acc[ti][tj], false, false);
    }

    // C/D layout: lane l -> n = l&15, m = (l>>4)*8 + r for VGPR r
    int n  = lane & 15;
    int mb = (lane >> 4) * 8;
    size_t cbase = (size_t)g * sCg;
#pragma unroll
    for (int ti = 0; ti < 4; ti++)
#pragma unroll
        for (int tj = 0; tj < 4; tj++)
#pragma unroll
            for (int r = 0; r < 8; r++) {
                size_t offc = cbase + (size_t)(row0 + ti * 16 + mb + r) * N
                                    + (size_t)(col0 + tj * 16 + n);
                if (OUT_BF16) ((bf16*)Cout)[offc] = f2bf(acc[ti][tj][r]);
                else          ((float*)Cout)[offc] = acc[ti][tj][r];
            }
}

// ---------------------------------------------------------------- instance-norm stats, stage 1: partial sums
__global__ void stats_partial_kernel(const float* __restrict__ S, float2* __restrict__ part,
                                     int d, float scale)
{
    __shared__ float ssum[256];
    __shared__ float ssq[256];
    int g = blockIdx.x;          // (b,h) in [0,32)
    int p = blockIdx.y;          // slice  in [0,8)
    int chunk = d * 120;         // (d*960)/8
    const float* base = S + (size_t)g * d * 960 + (size_t)p * chunk;
    float s = 0.f, q = 0.f;
    for (int i = threadIdx.x; i < chunk; i += 256) {
        float v = base[i] * scale;
        s += v; q += v * v;
    }
    ssum[threadIdx.x] = s; ssq[threadIdx.x] = q;
    __syncthreads();
    for (int st = 128; st > 0; st >>= 1) {
        if ((int)threadIdx.x < st) {
            ssum[threadIdx.x] += ssum[threadIdx.x + st];
            ssq[threadIdx.x]  += ssq[threadIdx.x + st];
        }
        __syncthreads();
    }
    if (threadIdx.x == 0) part[g * 8 + p] = make_float2(ssum[0], ssq[0]);
}

// stage 2: finalize mean / rstd per (b,h)
__global__ void stats_final_kernel(const float2* __restrict__ part, float2* __restrict__ stats, int d)
{
    int g = threadIdx.x;
    if (g >= 32) return;
    float s = 0.f, q = 0.f;
    for (int p = 0; p < 8; p++) { float2 v = part[g * 8 + p]; s += v.x; q += v.y; }
    float count = (float)d * 960.f;
    float mean = s / count;
    float var  = q / count - mean * mean;   // biased var, as InstanceNorm2d
    stats[g] = make_float2(mean, rsqrtf(var + 1e-5f));
}

// ---------------------------------------------------------------- fused norm + softmax row (len 960), writes bf16
__global__ void softmax_kernel(const float* __restrict__ S, const float2* __restrict__ stats,
                               bf16* __restrict__ Aout, int d, float scale)
{
    __shared__ float red[256];
    int row = blockIdx.x;           // 0 .. 32*d-1
    int g = row / d;
    const float* base = S + (size_t)row * 960;
    bf16* out = Aout + (size_t)row * 960;
    float2 st = stats[g];

    float x[4];
    int cnt = 0;
    float mx = -3.4e38f;
    for (int i = threadIdx.x, t = 0; i < 960; i += 256, t++) {
        float v = (base[i] * scale - st.x) * st.y;
        x[t] = v;
        if (v > mx) mx = v;
        cnt = t + 1;
    }
    red[threadIdx.x] = mx; __syncthreads();
    for (int s = 128; s > 0; s >>= 1) {
        if ((int)threadIdx.x < s) red[threadIdx.x] = fmaxf(red[threadIdx.x], red[threadIdx.x + s]);
        __syncthreads();
    }
    mx = red[0]; __syncthreads();

    float sum = 0.f;
    for (int t = 0; t < cnt; t++) { x[t] = __expf(x[t] - mx); sum += x[t]; }
    red[threadIdx.x] = sum; __syncthreads();
    for (int s = 128; s > 0; s >>= 1) {
        if ((int)threadIdx.x < s) red[threadIdx.x] += red[threadIdx.x + s];
        __syncthreads();
    }
    float inv = 1.0f / red[0];
    for (int i = threadIdx.x, t = 0; i < 960; i += 256, t++) out[i] = f2bf(x[t] * inv);
}

// ---------------------------------------------------------------- head mean + transpose via TDM
// Cf[b,h,e,n] (f32, n fastest) -> Cm[b,n,e] (bf16, e fastest).
// Each workgroup handles one (b, e-tile(64), n-tile(64)). Wave 0 issues 4 TDM
// tensor_load_to_lds ops (one 64x64 f32 tile per head) with LDS row padding of
// 1 dword per 64 dwords (pad_interval code 5, pad_amount code 0) -> 65-float
// rows, bank-conflict-free transposed reads. Then s_wait_tensorcnt 0 + barrier.
#define HM_ROW 65
#define HM_TILE (64 * HM_ROW)
__global__ __launch_bounds__(256) void headmean_kernel(const float* __restrict__ Cf,
                                                       bf16* __restrict__ Cm, int d)
{
    __shared__ float tile[4 * HM_TILE];   // single LDS object -> assumed at LDS offset 0

    int n0 = blockIdx.x * 64;
    int e0 = blockIdx.y * 64;
    int b  = blockIdx.z;

    if ((threadIdx.x >> 5) == 0) {        // wave 0 issues the DMAs
        for (int hh = 0; hh < 4; hh++) {
            const float* src = Cf + ((size_t)(b * 4 + hh) * d + e0) * 1024 + n0;
            unsigned long long ga = (unsigned long long)(uintptr_t)src;

            // D# group 0: count=1, lds_addr, global_addr[56:0], type=2
            v4u32 g0;
            g0.x = 1u;
            g0.y = (unsigned)(hh * HM_TILE * 4);
            g0.z = (unsigned)(ga & 0xffffffffull);
            g0.w = (unsigned)((ga >> 32) & 0x01ffffffull) | (2u << 30);

            // D# group 1: data_size=4B(code 2), pad_enable, pad_interval=64dw(code 5),
            // pad_amount=1dw(code 0); tensor_dim0=1024, tensor_dim1=d;
            // tile_dim0=64, tile_dim1=64; tensor_dim0_stride=1024.
            v8i32 g1;
            g1[0] = (int)((2u << 16) | (1u << 20) | (5u << 22));
            g1[1] = (int)(1024u << 16);          // tensor_dim0[15:0] @ bits 63:48
            g1[2] = (int)((unsigned)d << 16);    // tensor_dim1[15:0] @ bits 95:80
            g1[3] = (int)(64u << 16);            // tile_dim0 @ bits 127:112
            g1[4] = 64;                          // tile_dim1
            g1[5] = 1024;                        // tensor_dim0_stride[31:0]
            g1[6] = 0;
            g1[7] = 0;

            v4i32 z4 = {0, 0, 0, 0};             // groups 2/3 unused (2D tensor)
            v8i32 z8 = {0, 0, 0, 0, 0, 0, 0, 0}; // extra group (6-arg clang-23 form)
            __builtin_amdgcn_tensor_load_to_lds(g0, g1, z4, z4, z8, 0);
        }
        __builtin_amdgcn_s_wait_tensorcnt(0);
    }
    __syncthreads();

    int el = threadIdx.x & 63;               // e within tile (fastest on output)
    int nr = threadIdx.x >> 6;               // 0..3
    for (int nl = nr; nl < 64; nl += 4) {
        float s = tile[0 * HM_TILE + el * HM_ROW + nl]
                + tile[1 * HM_TILE + el * HM_ROW + nl]
                + tile[2 * HM_TILE + el * HM_ROW + nl]
                + tile[3 * HM_TILE + el * HM_ROW + nl];
        Cm[((size_t)b * 1024 + (n0 + nl)) * d + (e0 + el)] = f2bf(0.25f * s);
    }
}

// ---------------------------------------------------------------- launch
extern "C" void kernel_launch(void* const* d_in, const int* in_sizes, int n_in,
                              void* d_out, int out_size, void* d_ws, size_t ws_size,
                              hipStream_t stream)
{
    const int B = 8, N = 1024, H = 4, KV = 960;
    const float scale = 1.0f / sqrtf((float)KV);

    const float* emb[4] = { (const float*)d_in[0], (const float*)d_in[1],
                            (const float*)d_in[2], (const float*)d_in[3] };
    const float* emb_all = (const float*)d_in[4];
    const float* WQ[4]  = { (const float*)d_in[5], (const float*)d_in[6],
                            (const float*)d_in[7], (const float*)d_in[8] };
    const float* WK = (const float*)d_in[9];
    const float* WV = (const float*)d_in[10];
    const float* WO[4]  = { (const float*)d_in[11], (const float*)d_in[12],
                            (const float*)d_in[13], (const float*)d_in[14] };
    float* outp = (float*)d_out;

    // ---- workspace layout (sized for d=512, branch buffers reused)
    char* w = (char*)d_ws;
    size_t off = 0;
    auto take = [&](size_t bytes) -> void* {
        void* p = w + off;
        off = (off + bytes + 255) & ~(size_t)255;
        return p;
    };
    bf16*  eall_bf = (bf16*) take((size_t)B * N * KV * 2);        // 15.7 MB
    bf16*  wk_bf   = (bf16*) take((size_t)H * KV * KV * 2);       //  7.4 MB
    bf16*  wv_bf   = (bf16*) take((size_t)H * KV * KV * 2);       //  7.4 MB
    bf16*  kt_bf   = (bf16*) take((size_t)B * H * KV * N * 2);    // 62.9 MB  K^T [b,h,j,n]
    bf16*  v_bf    = (bf16*) take((size_t)B * H * N * KV * 2);    // 62.9 MB  V   [b,h,n,j]
    bf16*  emb_bf  = (bf16*) take((size_t)B * N * 512 * 2);       //  8.4 MB
    bf16*  wq_bf   = (bf16*) take((size_t)H * 512 * 512 * 2);     //  2.1 MB
    bf16*  wo_bf   = (bf16*) take((size_t)512 * 512 * 2);         //  0.5 MB
    bf16*  q_bf    = (bf16*) take((size_t)B * H * 512 * N * 2);   // 33.6 MB  Q [b,h,e,n]
    float* Xbuf    = (float*)take((size_t)B * H * 512 * N * 4);   // 67.1 MB  S then Cf
    bf16*  a_bf    = (bf16*) take((size_t)B * H * 512 * KV * 2);  // 31.5 MB  softmax(A)
    bf16*  cm_bf   = (bf16*) take((size_t)B * N * 512 * 2);       //  8.4 MB
    float2* partb  = (float2*)take((size_t)B * H * 8 * sizeof(float2));
    float2* statb  = (float2*)take((size_t)B * H * sizeof(float2));
    (void)ws_size; (void)n_in; (void)in_sizes; (void)out_size;

    auto cast = [&](const float* s, bf16* dstp, size_t n) {
        int blocks = (int)((n + 255) / 256);
        cast_kernel<<<blocks, 256, 0, stream>>>(s, dstp, n);
    };
    auto gemm = [&](bool outbf, const bf16* A, const bf16* Bp, void* C,
                    int M, int Nn, int Kk,
                    long sAi, long sAj, long sBi, long sBj, long sCg, int I, int J) {
        dim3 grid(Nn / 64, M / 64, I * J);
        if (outbf) gemm_bt<true ><<<grid, 32, 0, stream>>>(A, Bp, C, M, Nn, Kk, sAi, sAj, sBi, sBj, sCg, J);
        else       gemm_bt<false><<<grid, 32, 0, stream>>>(A, Bp, C, M, Nn, Kk, sAi, sAj, sBi, sBj, sCg, J);
    };

    // ---- shared K / V projections
    cast(emb_all, eall_bf, (size_t)B * N * KV);
    cast(WK, wk_bf, (size_t)H * KV * KV);
    cast(WV, wv_bf, (size_t)H * KV * KV);

    // K^T[b,h,j,n] = sum_k WK[h,j,k] * emb_all[b,n,k]
    gemm(true, wk_bf, eall_bf, kt_bf, KV, N, KV,
         0L, (long)KV * KV, (long)N * KV, 0L, (long)KV * N, B, H);
    // V[b,h,n,j] = sum_k emb_all[b,n,k] * WV[h,j,k]
    gemm(true, eall_bf, wv_bf, v_bf, N, KV, KV,
         (long)N * KV, 0L, 0L, (long)KV * KV, (long)N * KV, B, H);

    const int   dvals[4]  = { 64, 128, 256, 512 };
    size_t      outoff[4] = { 0,
                              (size_t)B * N * 64,
                              (size_t)B * N * (64 + 128),
                              (size_t)B * N * (64 + 128 + 256) };

    for (int br = 0; br < 4; br++) {
        const int d = dvals[br];

        cast(emb[br], emb_bf, (size_t)B * N * d);
        cast(WQ[br], wq_bf, (size_t)H * d * d);
        cast(WO[br], wo_bf, (size_t)d * d);

        // Q[b,h,e,n] = sum_k WQ[h,e,k] * emb[b,n,k]
        gemm(true, wq_bf, emb_bf, q_bf, d, N, d,
             0L, (long)d * d, (long)N * d, 0L, (long)d * N, B, H);

        // S[b,h,e,j] = sum_n Q[b,h,e,n] * K^T[b,h,j,n]   (f32, unscaled; scale folded into norm)
        gemm(false, q_bf, kt_bf, Xbuf, d, KV, N,
             (long)H * d * N, (long)d * N, (long)H * KV * N, (long)KV * N, (long)d * KV, B, H);

        // instance norm stats (two-stage) + fused norm/softmax -> bf16 A
        stats_partial_kernel<<<dim3(B * H, 8), 256, 0, stream>>>(Xbuf, partb, d, scale);
        stats_final_kernel<<<1, 32, 0, stream>>>(partb, statb, d);
        softmax_kernel<<<B * H * d, 256, 0, stream>>>(Xbuf, statb, a_bf, d, scale);

        // Cf[b,h,e,n] = sum_j A[b,h,e,j] * V[b,h,n,j]   (reuse Xbuf; S no longer needed)
        gemm(false, a_bf, v_bf, Xbuf, d, N, KV,
             (long)H * d * KV, (long)d * KV, (long)H * N * KV, (long)N * KV, (long)d * N, B, H);

        // head mean + transpose -> Cm[b,n,e] bf16 (TDM tiles into LDS)
        headmean_kernel<<<dim3(16, d / 64, B), 256, 0, stream>>>(Xbuf, cm_bf, d);

        // O[b,n,f] = sum_e Cm[b,n,e] * WO[f,e]  -> f32 directly into d_out slice
        gemm(false, cm_bf, wo_bf, outp + outoff[br], N, d, d,
             (long)N * d, 0L, 0L, 0L, (long)N * d, B, 1);
    }
}
